// MultiHeadAttention_61529701482668
// MI455X (gfx1250) — matrix operand order repaired and checked by reading, not compiled, not run on hardware
//
#include <hip/hip_runtime.h>

// ---------------------------------------------------------------------------
// Problem constants (match reference setup_inputs)
// ---------------------------------------------------------------------------
#define BB   4
#define SS   2048
#define EE   2048
#define HH   16
#define KVHH 4
#define DD   128
#define GG   (HH / KVHH)      // 4
#define KVD  (KVHH * DD)      // 512
#define MM   (BB * SS)        // 8192

typedef __bf16 bf16_t;
typedef __attribute__((ext_vector_type(16))) __bf16 v16bf;
typedef __attribute__((ext_vector_type(8)))  __bf16 v8bf;
typedef __attribute__((ext_vector_type(8)))  float  v8f;
typedef __attribute__((ext_vector_type(4)))  float  v4f;

// Build a 16-element bf16 fragment from two contiguous 16-byte chunks.
// Used with contiguous k order (k = 16*(lane/16) + s) for both A and B (elements ascend in k within a lane).
static __device__ __forceinline__ v16bf frag16(const bf16_t* p0, const bf16_t* p1) {
  v8bf lo = *(const v8bf*)p0;
  v8bf hi = *(const v8bf*)p1;
  return __builtin_shufflevector(lo, hi, 0,1,2,3,4,5,6,7,8,9,10,11,12,13,14,15);
}

static __device__ __forceinline__ v8f wmma_bf16(v16bf a, v16bf b, v8f c) {
  // D = A(16x32) * B(32x16) + C, f32 accumulate
  return __builtin_amdgcn_wmma_f32_16x16x32_bf16(false, a, false, b, (short)0, c, false, false);
}

// ---------------------------------------------------------------------------
// fp32 -> bf16 conversion, 8 elements/thread (b128 in/out)
// ---------------------------------------------------------------------------
__global__ void cvt_f32_to_bf16(const float* __restrict__ in, bf16_t* __restrict__ out, int n8) {
  int i = blockIdx.x * blockDim.x + threadIdx.x;
  if (i >= n8) return;
  const v4f a = ((const v4f*)in)[2 * i];
  const v4f b = ((const v4f*)in)[2 * i + 1];
  v8bf o;
  o[0] = (bf16_t)a[0]; o[1] = (bf16_t)a[1]; o[2] = (bf16_t)a[2]; o[3] = (bf16_t)a[3];
  o[4] = (bf16_t)b[0]; o[5] = (bf16_t)b[1]; o[6] = (bf16_t)b[2]; o[7] = (bf16_t)b[3];
  ((v8bf*)out)[i] = o;
}

// ---------------------------------------------------------------------------
// GEMM: C[M,N] = A[M,K](bf16) @ W[N,K](bf16)^T + bias[N]
// Block: 256 threads = 8 waves, block tile 128x128, wave tile 32x64.
// A-fragment: row = lane%16, 16 contiguous k at k-base = (lane/16)*16.
// B-fragment: n = lane%16, 16 contiguous k at k-base = (lane/16)*16 (w row-major).
// ---------------------------------------------------------------------------
template <int OUT_BF16>
__global__ __launch_bounds__(256) void gemm_bias_wmma(
    const bf16_t* __restrict__ A, const bf16_t* __restrict__ W,
    const float* __restrict__ bias, void* __restrict__ out,
    int M, int N, int K)
{
  const int lane = threadIdx.x & 31;
  const int wave = threadIdx.x >> 5;
  const int lr = lane & 15;
  const int lh = lane >> 4;
  const int mb = blockIdx.x * 128 + (wave & 3) * 32;   // 4 waves along M
  const int nb = blockIdx.y * 128 + (wave >> 2) * 64;  // 2 waves along N

  v8f acc[2][4];
#pragma unroll
  for (int i = 0; i < 2; ++i)
#pragma unroll
    for (int j = 0; j < 4; ++j) acc[i][j] = {};

  const bf16_t* ap0 = A + (size_t)(mb + lr) * K + lh * 16;
  const bf16_t* ap1 = ap0 + (size_t)16 * K;
  const bf16_t* wp  = W + (size_t)(nb + lr) * K + lh * 16;

  for (int k = 0; k < K; k += 32) {
    v16bf af[2], wf[4];
    af[0] = frag16(ap0 + k, ap0 + k + 8);
    af[1] = frag16(ap1 + k, ap1 + k + 8);
    __builtin_prefetch(ap0 + k + 64, 0, 1);
#pragma unroll
    for (int j = 0; j < 4; ++j) {
      const bf16_t* p = wp + (size_t)j * 16 * K + k;
      wf[j] = frag16(p, p + 8);
    }
#pragma unroll
    for (int i = 0; i < 2; ++i)
#pragma unroll
      for (int j = 0; j < 4; ++j)
        acc[i][j] = wmma_bf16(af[i], wf[j], acc[i][j]);
  }

  // Epilogue: C layout -> VGPR r holds row (r + 8*(lane/16)), col = lane%16.
#pragma unroll
  for (int i = 0; i < 2; ++i)
#pragma unroll
    for (int j = 0; j < 4; ++j) {
      const int col = nb + j * 16 + lr;
      const float bv = bias[col];
#pragma unroll
      for (int r = 0; r < 8; ++r) {
        const int row = mb + i * 16 + r + lh * 8;
        float v = acc[i][j][r] + bv;
        if (OUT_BF16) ((bf16_t*)out)[(size_t)row * N + col] = (bf16_t)v;
        else          ((float*)out)[(size_t)row * N + col] = v;
      }
    }
}

// ---------------------------------------------------------------------------
// RoPE + [B,S,Hn,D] -> [B,Hn,S,D] transpose (scale folds 1/sqrt(D) into Q)
// ---------------------------------------------------------------------------
__global__ void rope_transpose_kernel(const bf16_t* __restrict__ in, bf16_t* __restrict__ out,
                                      int Hn, float scale, int total)
{
  int idx = blockIdx.x * blockDim.x + threadIdx.x;
  if (idx >= total) return;
  int t = idx;
  const int d2 = t & 63;           t >>= 6;   // D/2 = 64
  const int s  = t & (SS - 1);     t >>= 11;
  const int h  = t % Hn;
  const int b  = t / Hn;
  // inv_freq = 10000^(-d2/64) = exp(-d2 * ln(10000)/64)
  const float ang = (float)s * __expf(-(float)d2 * 0.14391565f);
  float c, sn;
  __sincosf(ang, &c, &sn);
  const size_t ib = (((size_t)b * SS + s) * Hn + h) * DD;
  const float v0 = (float)in[ib + d2];
  const float v1 = (float)in[ib + d2 + 64];
  const size_t ob = (((size_t)b * Hn + h) * SS + s) * DD;
  out[ob + d2]      = (bf16_t)((v0 * c - v1 * sn) * scale);
  out[ob + d2 + 64] = (bf16_t)((v1 * c + v0 * sn) * scale);
}

// V: [B,S,KVH,D] -> [B,KVH,D,S]  (keys contiguous, needed by PV B-fragments)
__global__ void v_transpose_kernel(const bf16_t* __restrict__ in, bf16_t* __restrict__ out)
{
  int t = blockIdx.x * blockDim.x + threadIdx.x;
  const int d  = t & 127;          t >>= 7;
  const int kv = t & (KVHH - 1);   t >>= 2;
  const int s  = t & (SS - 1);     t >>= 11;
  const int b  = t;
  out[(((size_t)b * KVHH + kv) * DD + d) * SS + s] =
      in[(((size_t)b * SS + s) * KVHH + kv) * DD + d];
}

// ---------------------------------------------------------------------------
// Flash attention (causal, GQA). One wave owns 16 query rows.
// Processes 64 keys per step; MASKED=false path (the common case) has no
// causal cndmask chains at all.
// ---------------------------------------------------------------------------
template <bool MASKED>
static __device__ __forceinline__ void attn_block64(
    const bf16_t* __restrict__ kp, const bf16_t* __restrict__ vp,
    bf16_t* __restrict__ pb, const v16bf (&qf)[4],
    v8f (&acc)[8], float (&mrow)[8], float (&lrow)[8],
    const int kbase, const int q0, const int lr, const int lh)
{
  // S-tile: [16 q x 64 keys] = four 16x16 accumulators, 16 WMMA over d
  v8f s[4] = {{}, {}, {}, {}};
#pragma unroll
  for (int dc = 0; dc < 4; ++dc) {
#pragma unroll
    for (int ti = 0; ti < 4; ++ti) {
      const bf16_t* p = kp + (size_t)(kbase + ti * 16 + lr) * DD + dc * 32 + lh * 16;
      s[ti] = wmma_bf16(qf[dc], frag16(p, p + 8), s[ti]);
    }
  }

  // Online softmax; P written to LDS in row-major [16 x 64]
#pragma unroll
  for (int r = 0; r < 8; ++r) {
    const int qrow = q0 + r + lh * 8;
    float x[4];
#pragma unroll
    for (int ti = 0; ti < 4; ++ti) {
      x[ti] = s[ti][r];
      if (MASKED) x[ti] = (kbase + ti * 16 + lr <= qrow) ? x[ti] : -1e30f;
    }
    float mx = fmaxf(fmaxf(x[0], x[1]), fmaxf(x[2], x[3]));
#pragma unroll
    for (int off = 8; off; off >>= 1) mx = fmaxf(mx, __shfl_xor(mx, off, 32));
    const float mnew  = fmaxf(mrow[r], mx);
    const float alpha = __expf(mrow[r] - mnew);
    float p[4], sum = 0.0f;
#pragma unroll
    for (int ti = 0; ti < 4; ++ti) { p[ti] = __expf(x[ti] - mnew); sum += p[ti]; }
#pragma unroll
    for (int off = 8; off; off >>= 1) sum += __shfl_xor(sum, off, 32);
    lrow[r] = lrow[r] * alpha + sum;
    mrow[r] = mnew;
#pragma unroll
    for (int t = 0; t < 8; ++t) acc[t][r] *= alpha;
    const int prow = r + lh * 8;
#pragma unroll
    for (int ti = 0; ti < 4; ++ti)
      pb[prow * 64 + ti * 16 + lr] = (bf16_t)p[ti];
  }

  // Per-wave LDS ops are in-order; wait for stores before fragment reload.
  asm volatile("s_wait_dscnt 0x0" ::: "memory");

  // Reload P as two A-fragments (k = 0..31 and 32..63)
  const bf16_t* pp = pb + lr * 64 + lh * 16;
  const v16bf pf0 = frag16(pp, pp + 8);
  const v16bf pf1 = frag16(pp + 32, pp + 40);

  // O += P(16x64) @ V(64x128): 8 d-tiles x 2 key-halves, keys contiguous in Vt
#pragma unroll
  for (int t = 0; t < 8; ++t) {
    const bf16_t* vb = vp + (size_t)(t * 16 + lr) * SS + kbase + lh * 16;
    acc[t] = wmma_bf16(pf0, frag16(vb, vb + 8), acc[t]);
    acc[t] = wmma_bf16(pf1, frag16(vb + 32, vb + 40), acc[t]);
  }
}

__global__ __launch_bounds__(256) void flash_attn_wmma(
    const bf16_t* __restrict__ Q, const bf16_t* __restrict__ Km,
    const bf16_t* __restrict__ Vt, bf16_t* __restrict__ O)
{
  __shared__ bf16_t pbuf[8][16 * 64];   // per-wave P tile (16 q-rows x 64 keys)
  const int lane = threadIdx.x & 31;
  const int wave = threadIdx.x >> 5;
  const int lr = lane & 15;
  const int lh = lane >> 4;
  const int b   = blockIdx.x / HH;
  const int h   = blockIdx.x % HH;
  const int kvh = h / GG;                       // contiguous-repeat GQA mapping
  const int q0  = blockIdx.y * 128 + wave * 16; // this wave's query-row base

  const bf16_t* qp = Q  + ((size_t)b * HH   + h)   * SS * DD;
  const bf16_t* kp = Km + ((size_t)b * KVHH + kvh) * SS * DD;
  const bf16_t* vp = Vt + ((size_t)b * KVHH + kvh) * DD * SS;
  bf16_t* pb = &pbuf[wave][0];

  // Preload Q A-fragments for the 4 d-chunks of 32 (reused across all key blocks)
  v16bf qf[4];
#pragma unroll
  for (int dc = 0; dc < 4; ++dc) {
    const bf16_t* p = qp + (size_t)(q0 + lr) * DD + dc * 32 + lh * 16;
    qf[dc] = frag16(p, p + 8);
  }

  v8f acc[8];                 // O accumulator: 16 rows x 128 cols
  float mrow[8], lrow[8];     // online-softmax stats; index r -> row (r + 8*lh)
#pragma unroll
  for (int t = 0; t < 8; ++t) acc[t] = {};
#pragma unroll
  for (int r = 0; r < 8; ++r) { mrow[r] = -1e30f; lrow[r] = 0.0f; }

  const int nfull = q0 >> 6;          // blocks with kbase+63 <= q0: no masking
  const int nkb   = (q0 + 79) >> 6;   // total blocks covering keys <= q0+15
  int kb = 0;
  for (; kb < nfull; ++kb)
    attn_block64<false>(kp, vp, pb, qf, acc, mrow, lrow, kb * 64, q0, lr, lh);
  for (; kb < nkb; ++kb)
    attn_block64<true>(kp, vp, pb, qf, acc, mrow, lrow, kb * 64, q0, lr, lh);

  // Normalize and store to [B,S,H*D]
#pragma unroll
  for (int t = 0; t < 8; ++t) {
    const int d = t * 16 + lr;
#pragma unroll
    for (int r = 0; r < 8; ++r) {
      const int qrow = q0 + r + lh * 8;
      O[((size_t)b * SS + qrow) * EE + h * DD + d] = (bf16_t)(acc[t][r] / lrow[r]);
    }
  }
}

// ---------------------------------------------------------------------------
// Launcher
// ---------------------------------------------------------------------------
extern "C" void kernel_launch(void* const* d_in, const int* in_sizes, int n_in,
                              void* d_out, int out_size, void* d_ws, size_t ws_size,
                              hipStream_t stream)
{
  const float* x  = (const float*)d_in[0];
  const float* wq = (const float*)d_in[1];
  const float* bq = (const float*)d_in[2];
  const float* wk = (const float*)d_in[3];
  const float* bk = (const float*)d_in[4];
  const float* wv = (const float*)d_in[5];
  const float* bv = (const float*)d_in[6];
  const float* wo = (const float*)d_in[7];
  const float* bo = (const float*)d_in[8];

  char* w = (char*)d_ws;
  auto carve = [&](size_t bytes) -> void* {
    void* p = (void*)w;
    w += (bytes + 255) & ~(size_t)255;
    return p;
  };

  bf16_t* xb  = (bf16_t*)carve((size_t)MM * EE * 2);   // x bf16
  bf16_t* wqb = (bf16_t*)carve((size_t)EE * EE * 2);
  bf16_t* wkb = (bf16_t*)carve((size_t)KVD * EE * 2);
  bf16_t* wvb = (bf16_t*)carve((size_t)KVD * EE * 2);
  bf16_t* wob = (bf16_t*)carve((size_t)EE * EE * 2);
  bf16_t* qpb = (bf16_t*)carve((size_t)MM * EE * 2);   // q proj [B,S,H,D]
  bf16_t* kpb = (bf16_t*)carve((size_t)MM * KVD * 2);  // k proj [B,S,KV,D]
  bf16_t* vpb = (bf16_t*)carve((size_t)MM * KVD * 2);  // v proj [B,S,KV,D]
  bf16_t* qb  = (bf16_t*)carve((size_t)MM * EE * 2);   // q rope [B,H,S,D]
  bf16_t* kbf = (bf16_t*)carve((size_t)MM * KVD * 2);  // k rope [B,KV,S,D]
  bf16_t* vtb = (bf16_t*)carve((size_t)MM * KVD * 2);  // v^T    [B,KV,D,S]
  bf16_t* ob  = (bf16_t*)carve((size_t)MM * EE * 2);   // attn out [B,S,E]

  auto cvt = [&](const float* src, bf16_t* dst, size_t n) {
    const int n8 = (int)(n / 8);
    cvt_f32_to_bf16<<<(unsigned)((n8 + 255) / 256), 256, 0, stream>>>(src, dst, n8);
  };
  cvt(x,  xb,  (size_t)MM * EE);
  cvt(wq, wqb, (size_t)EE * EE);
  cvt(wk, wkb, (size_t)KVD * EE);
  cvt(wv, wvb, (size_t)KVD * EE);
  cvt(wo, wob, (size_t)EE * EE);

  const dim3 blk(256);
  gemm_bias_wmma<1><<<dim3(MM / 128, EE  / 128), blk, 0, stream>>>(xb, wqb, bq, qpb, MM, EE,  EE);
  gemm_bias_wmma<1><<<dim3(MM / 128, KVD / 128), blk, 0, stream>>>(xb, wkb, bk, kpb, MM, KVD, EE);
  gemm_bias_wmma<1><<<dim3(MM / 128, KVD / 128), blk, 0, stream>>>(xb, wvb, bv, vpb, MM, KVD, EE);

  const float qscale = 0.08838834764831845f;  // 1/sqrt(128), folded into Q
  {
    const int total = BB * HH * SS * (DD / 2);
    rope_transpose_kernel<<<total / 256, 256, 0, stream>>>(qpb, qb, HH, qscale, total);
  }
  {
    const int total = BB * KVHH * SS * (DD / 2);
    rope_transpose_kernel<<<total / 256, 256, 0, stream>>>(kpb, kbf, KVHH, 1.0f, total);
  }
  {
    const int total = BB * SS * KVHH * DD;
    v_transpose_kernel<<<total / 256, 256, 0, stream>>>(vpb, vtb);
  }

  flash_attn_wmma<<<dim3(BB * HH, SS / 128), blk, 0, stream>>>(qb, kbf, vtb, ob);

  gemm_bias_wmma<0><<<dim3(MM / 128, EE / 128), blk, 0, stream>>>(ob, wob, bo, d_out, MM, EE, EE);
}
